// DMGI_80908593922177
// MI455X (gfx1250) — compile-verified, matrix operand order checked by reading, not run on hardware
//
#include <hip/hip_runtime.h>

#define NN   50000
#define EE   1600000
#define RR   4
#define FIN  128
#define FOUT 64
#define LDSP 136   // padded LDS row pitch in f16 elements (avoids 16-way bank conflicts)

typedef __attribute__((ext_vector_type(16))) _Float16 v16h;
typedef __attribute__((ext_vector_type(8)))  _Float16 v8h;
typedef __attribute__((ext_vector_type(4)))  _Float16 v4h;
typedef __attribute__((ext_vector_type(8)))  float    v8f;

// ---------------------------------------------------------------------------
// Zero deg accumulators (ws) and summaries (d_out tail) once per call.
// ---------------------------------------------------------------------------
__global__ void dmgi_zero(float* __restrict__ deg, float* __restrict__ summ) {
    int idx = blockIdx.x * blockDim.x + threadIdx.x;
    if (idx < RR * NN) deg[idx] = 0.0f;
    if (idx < RR * 64) summ[idx] = 0.0f;
}

// ---------------------------------------------------------------------------
// deg[r][dst] += 1 over all relations' edges (non-returning f32 atomics).
// ---------------------------------------------------------------------------
__global__ void dmgi_deg(const int* __restrict__ ei, float* __restrict__ deg) {
    long idx = (long)blockIdx.x * blockDim.x + threadIdx.x;
    if (idx >= (long)RR * EE) return;
    int r = (int)(idx / EE);
    int e = (int)(idx % EE);
    int d = ei[(long)r * 2 * EE + EE + e];   // ei[r][1][e]
    atomicAdd(&deg[r * NN + d], 1.0f);
}

// dinv = rsqrt(deg + 1)   (+1 = self loop)
__global__ void dmgi_dinv(const float* __restrict__ deg, float* __restrict__ dinv) {
    int idx = blockIdx.x * blockDim.x + threadIdx.x;
    if (idx < RR * NN) dinv[idx] = rsqrtf(deg[idx] + 1.0f);
}

// ---------------------------------------------------------------------------
// Pre-pack W[r] (128x64 f32, row major) into WMMA B-fragment order, f16.
// Fragment fi = ks*4+nt; lane l holds col n = nt*16 + (l%16),
// K elements {k0..k0+7} U {k0+16..k0+23}, k0 = ks*32 + (l/16)*8.
// Bpack layout: [r][fi][lane][16] -> lane's v16h is one contiguous 32B chunk.
// ---------------------------------------------------------------------------
__global__ void dmgi_prepack(const float* __restrict__ W, _Float16* __restrict__ Bpack) {
    int idx = blockIdx.x * blockDim.x + threadIdx.x;
    if (idx >= RR * 16 * 32 * 16) return;
    int j    = idx & 15;
    int lane = (idx >> 4) & 31;
    int fi   = (idx >> 9) & 15;
    int r    = idx >> 13;
    int ks = fi >> 2, nt = fi & 3;
    int k = ks * 32 + ((lane >> 4) * 8) + (j < 8 ? j : j + 8);
    int n = nt * 16 + (lane & 15);
    Bpack[idx] = (_Float16)W[(long)r * FIN * FOUT + k * FOUT + n];
}

// ---------------------------------------------------------------------------
// xw = relu-free projection: ((x .* mask)[perm]) @ W  via v_wmma_f32_16x16x32_f16
// Block = 128 threads = 4 waves; each wave computes a 16x64 strip.
// A tile (64 rows x 128 cols f16) staged in LDS with fused mask multiply and
// optional row-permutation gather.  B fragments live in registers (16 x v16h).
// ---------------------------------------------------------------------------
__global__ __launch_bounds__(128) void dmgi_gemm(
    const float* __restrict__ x,
    const float* __restrict__ mask,      // per-relation [N,128]
    const int*   __restrict__ perm,      // nullptr for pos path
    const _Float16* __restrict__ Bpack,  // per-relation packed fragments
    float* __restrict__ xw)              // [N,64]
{
    __shared__ _Float16 lds[64 * LDSP];
    const int t = threadIdx.x;
    const int rowBase = blockIdx.x * 64;

    // ---- cooperative coalesced load: 4 rows/pass x 16 passes, float4 wide ----
    {
        const int rl = t >> 5;          // 0..3
        const int c  = (t & 31) * 4;    // 0..124
        for (int pass = 0; pass < 16; ++pass) {
            int i = pass * 4 + rl;      // local row 0..63
            int row = rowBase + i;
            if (row >= NN) row = NN - 1;           // clamp (results unused)
            int src = perm ? perm[row] : row;
            const float4 xv = *(const float4*)(x    + (long)src * FIN + c);
            const float4 mv = *(const float4*)(mask + (long)src * FIN + c);
            v4h h;
            h.x = (_Float16)(xv.x * mv.x);
            h.y = (_Float16)(xv.y * mv.y);
            h.z = (_Float16)(xv.z * mv.z);
            h.w = (_Float16)(xv.w * mv.w);
            *(v4h*)&lds[i * LDSP + c] = h;
        }
    }
    __syncthreads();

    const int lane = t & 31;
    const int wrow = (t >> 5) * 16;      // wave's 16-row strip inside block
    const int m    = lane & 15;
    const int k0   = (lane >> 4) * 8;

    // ---- B fragments: 16 contiguous 32B loads, kept in VGPRs ----
    v16h bfrag[16];
#pragma unroll
    for (int fi = 0; fi < 16; ++fi)
        bfrag[fi] = *(const v16h*)(Bpack + ((fi * 32 + lane) << 4));

    v8f c0 = {}, c1 = {}, c2 = {}, c3 = {};
#pragma unroll
    for (int ks = 0; ks < 4; ++ks) {
        const _Float16* ap = &lds[(wrow + m) * LDSP + ks * 32 + k0];
        v8h alo = *(const v8h*)(ap);        // K = k0..k0+7       (16B ds_load)
        v8h ahi = *(const v8h*)(ap + 16);   // K = k0+16..k0+23
        v16h a = __builtin_shufflevector(alo, ahi,
                 0,1,2,3,4,5,6,7,8,9,10,11,12,13,14,15);
        c0 = __builtin_amdgcn_wmma_f32_16x16x32_f16(false, a, false, bfrag[ks*4+0], (short)0, c0, false, false);
        c1 = __builtin_amdgcn_wmma_f32_16x16x32_f16(false, a, false, bfrag[ks*4+1], (short)0, c1, false, false);
        c2 = __builtin_amdgcn_wmma_f32_16x16x32_f16(false, a, false, bfrag[ks*4+2], (short)0, c2, false, false);
        c3 = __builtin_amdgcn_wmma_f32_16x16x32_f16(false, a, false, bfrag[ks*4+3], (short)0, c3, false, false);
    }

    // ---- C layout: lane -> col n=m, VGPR v -> row (lane/16)*8 + v ----
#pragma unroll
    for (int v = 0; v < 8; ++v) {
        int row = rowBase + wrow + (lane >> 4) * 8 + v;
        if (row < NN) {
            float* o = xw + (long)row * FOUT;
            o[ 0 + m] = c0[v];
            o[16 + m] = c1[v];
            o[32 + m] = c2[v];
            o[48 + m] = c3[v];
        }
    }
}

// ---------------------------------------------------------------------------
// Self-loop init straight into d_out: agg = xw * dinv^2 (pos & neg fused).
// ---------------------------------------------------------------------------
__global__ void dmgi_init(const float* __restrict__ xw_pos,
                          const float* __restrict__ xw_neg,
                          const float* __restrict__ dinv_r,
                          float* __restrict__ out_pos,
                          float* __restrict__ out_neg) {
    long idx = (long)blockIdx.x * blockDim.x + threadIdx.x;
    if (idx >= (long)NN * FOUT) return;
    int i = (int)(idx >> 6);
    float di = dinv_r[i];
    float c = di * di;
    out_pos[idx] = xw_pos[idx] * c;
    out_neg[idx] = xw_neg[idx] * c;
}

// ---------------------------------------------------------------------------
// Edge gather + scatter-add, 64 threads per edge (256B granularity),
// pos & neg convs share one pass over indices/coefs.
// ---------------------------------------------------------------------------
__global__ __launch_bounds__(256) void dmgi_scatter(
    const int*   __restrict__ ei_r,     // [2,E]
    const float* __restrict__ dinv_r,
    const float* __restrict__ xw_pos,
    const float* __restrict__ xw_neg,
    float* __restrict__ out_pos,
    float* __restrict__ out_neg)
{
    long idx = (long)blockIdx.x * blockDim.x + threadIdx.x;
    if (idx >= (long)EE * FOUT) return;
    int e = (int)(idx >> 6);
    int f = (int)(idx & 63);
    int s = ei_r[e];
    int d = ei_r[EE + e];
    float c = dinv_r[s] * dinv_r[d];
    atomicAdd(out_pos + (long)d * FOUT + f, xw_pos[(long)s * FOUT + f] * c);
    atomicAdd(out_neg + (long)d * FOUT + f, xw_neg[(long)s * FOUT + f] * c);
}

// ---------------------------------------------------------------------------
// In-place bias + ReLU for pos & neg, plus block-reduced column mean of pos_h.
// ---------------------------------------------------------------------------
__global__ __launch_bounds__(256) void dmgi_finalize(
    float* __restrict__ out_pos, float* __restrict__ out_neg,
    const float* __restrict__ b_r, float* __restrict__ summ_r)
{
    __shared__ float red[256];
    int f = threadIdx.x & 63;
    int g = threadIdx.x >> 6;            // 0..3
    float bias = b_r[f];
    float sum = 0.0f;
    for (long i = (long)blockIdx.x * 4 + g; i < NN; i += (long)gridDim.x * 4) {
        long idx = i * FOUT + f;
        float vp = out_pos[idx] + bias; vp = vp > 0.0f ? vp : 0.0f; out_pos[idx] = vp;
        float vn = out_neg[idx] + bias; vn = vn > 0.0f ? vn : 0.0f; out_neg[idx] = vn;
        sum += vp;
    }
    red[threadIdx.x] = sum;
    __syncthreads();
    if (g == 0) {
        float s = red[f] + red[64 + f] + red[128 + f] + red[192 + f];
        atomicAdd(summ_r + f, s * (1.0f / NN));
    }
}

// ---------------------------------------------------------------------------
extern "C" void kernel_launch(void* const* d_in, const int* in_sizes, int n_in,
                              void* d_out, int out_size, void* d_ws, size_t ws_size,
                              hipStream_t stream) {
    const float* x        = (const float*)d_in[0];
    const int*   ei       = (const int*)  d_in[1];   // [R,2,E]
    const float* W        = (const float*)d_in[2];   // [R,128,64]
    const float* b        = (const float*)d_in[3];   // [R,64]
    const float* pos_mask = (const float*)d_in[4];   // [R,N,128]
    const float* neg_mask = (const float*)d_in[5];   // [R,N,128]
    const int*   perm     = (const int*)  d_in[6];   // [R,N]

    // workspace carve-up (float-granular; Bpack lands 32B aligned: 27.2e6 B)
    float* ws      = (float*)d_ws;
    float* xw_pos  = ws;                              // N*64
    float* xw_neg  = xw_pos + (long)NN * FOUT;        // N*64
    float* deg     = xw_neg + (long)NN * FOUT;        // R*N
    float* dinv    = deg + (long)RR * NN;             // R*N
    _Float16* Bpack = (_Float16*)(dinv + (long)RR * NN);  // R*8192 f16

    float* out     = (float*)d_out;
    float* out_pos = out;                                     // [R,N,64]
    float* out_neg = out + (long)RR * NN * FOUT;              // [R,N,64]
    float* summ    = out + 2L * RR * NN * FOUT;               // [R,64]

    const int B = 256;

    // degree / normalization (all relations at once) + zero summaries
    dmgi_zero<<<(RR * NN + B - 1) / B, B, 0, stream>>>(deg, summ);
    dmgi_deg <<<(int)(((long)RR * EE + B - 1) / B), B, 0, stream>>>(ei, deg);
    dmgi_dinv<<<(RR * NN + B - 1) / B, B, 0, stream>>>(deg, dinv);

    // pack W into WMMA B-fragment layout (f16)
    dmgi_prepack<<<(RR * 16 * 32 * 16 + B - 1) / B, B, 0, stream>>>(W, Bpack);

    const int gemm_grid  = (NN + 63) / 64;                      // 782
    const int initn_grid = (int)(((long)NN * FOUT + B - 1) / B);
    const int scat_grid  = (int)(((long)EE * FOUT + B - 1) / B);

    for (int r = 0; r < RR; ++r) {
        const float*  pm   = pos_mask + (long)r * NN * FIN;
        const float*  nm   = neg_mask + (long)r * NN * FIN;
        const int*    pr   = perm + (long)r * NN;
        const int*    eir  = ei + (long)r * 2 * EE;
        const float*  dvr  = dinv + (long)r * NN;
        const _Float16* Bp = Bpack + (long)r * 8192;
        float* opr = out_pos + (long)r * NN * FOUT;
        float* onr = out_neg + (long)r * NN * FOUT;

        // projections (mask / permute fused into A-tile load)
        dmgi_gemm<<<gemm_grid, 128, 0, stream>>>(x, pm, nullptr, Bp, xw_pos);
        dmgi_gemm<<<gemm_grid, 128, 0, stream>>>(x, nm, pr,      Bp, xw_neg);

        // self-loop init directly in d_out
        dmgi_init<<<initn_grid, B, 0, stream>>>(xw_pos, xw_neg, dvr, opr, onr);

        // edge gather + atomic scatter (pos & neg in one pass)
        dmgi_scatter<<<scat_grid, B, 0, stream>>>(eir, dvr, xw_pos, xw_neg, opr, onr);

        // bias + relu + pos column mean
        dmgi_finalize<<<256, B, 0, stream>>>(opr, onr, b + r * 64, summ + r * 64);
    }
}
